// MultiheadDiffAttn_85779086835744
// MI455X (gfx1250) — matrix operand order repaired and checked by reading, not compiled, not run on hardware
//
#include <hip/hip_runtime.h>
#include <hip/hip_bf16.h>

#define B_ 2
#define T_ 2048
#define E_ 1024
#define H_ 16
#define D_ 64
#define LAMBDA_INIT 0.4707130183122507f
#define ONE_MINUS_LI 0.5292869816877493f
#define EPS_ 1e-5f

typedef _Float16 h8  __attribute__((ext_vector_type(8)));
typedef _Float16 h16 __attribute__((ext_vector_type(16)));
typedef float    f8  __attribute__((ext_vector_type(8)));

__device__ __forceinline__ f8 wmma_f16(h16 a, h16 b, f8 c) {
  return __builtin_amdgcn_wmma_f32_16x16x32_f16(false, a, false, b, (short)0, c,
                                                false, false);
}

__device__ __forceinline__ f8 zero8() {
  f8 z;
#pragma unroll
  for (int i = 0; i < 8; ++i) z[i] = 0.f;
  return z;
}

// A fragment 16x32 f16 (MxK). Lanes 0-15: rows 0-15, K range kh..kh+15
// with kh = 16*(lane>>4).
__device__ __forceinline__ h16 load_a_frag(const _Float16* base, int ld, int m0,
                                           int k0, int lane) {
  int m  = m0 + (lane & 15);
  int kh = (lane >> 4) << 4;
  const _Float16* p = base + (size_t)m * ld + k0 + kh;
  h8 lo = *(const h8*)(p);
  h8 hi = *(const h8*)(p + 8);
  h16 r;
#pragma unroll
  for (int i = 0; i < 8; ++i) { r[i] = lo[i]; r[i + 8] = hi[i]; }
  return r;
}

// B fragment 32x16 f16 (KxN), N-major storage (row n of `base` is contiguous in K).
// lane holds column n = n0 + (lane&15), K range k0 + 16*(lane>>4) .. +15.
__device__ __forceinline__ h16 load_b_frag(const _Float16* base, int ld, int n0,
                                           int k0, int lane) {
  int n  = n0 + (lane & 15);
  int kb = k0 + ((lane >> 4) << 4);
  return *(const h16*)(base + (size_t)n * ld + kb);
}

__global__ void cast_f32_f16(const float* __restrict__ src,
                             _Float16* __restrict__ dst, long n) {
  long i = (long)blockIdx.x * blockDim.x + threadIdx.x;
  long stride = (long)gridDim.x * blockDim.x;
  for (; i < n; i += stride) dst[i] = (_Float16)src[i];
}

__global__ void lam_kernel(const float* __restrict__ lq1, const float* __restrict__ lk1,
                           const float* __restrict__ lq2, const float* __restrict__ lk2,
                           float* __restrict__ lam) {
  int lane = threadIdx.x;  // 32 threads, D=64 -> 2 elements each
  float p1 = lq1[lane] * lk1[lane] + lq1[lane + 32] * lk1[lane + 32];
  float p2 = lq2[lane] * lk2[lane] + lq2[lane + 32] * lk2[lane + 32];
#pragma unroll
  for (int off = 16; off >= 1; off >>= 1) {
    p1 += __shfl_xor(p1, off, 32);
    p2 += __shfl_xor(p2, off, 32);
  }
  if (lane == 0) *lam = __expf(p1) - __expf(p2) + LAMBDA_INIT;
}

// C = A[M,K] x W[N,K]^T. 256 threads = 8 waves (2x4), 64x128 per block, 32x32 per wave.
// mode 0: f16 head layout [B,H,T,D] (scaled)   mode 1: f16 V^T layout [B,H,D,T]
// mode 2: f32 plain [M,N] (final output)
__global__ __launch_bounds__(256) void gemm_wmma(
    const _Float16* __restrict__ A, const _Float16* __restrict__ W,
    _Float16* __restrict__ out16, float* __restrict__ out32,
    int M, int N, int K, int mode, float scale) {
  int lane  = threadIdx.x & 31;
  int wave  = threadIdx.x >> 5;
  int mBase = blockIdx.x * 64 + (wave & 1) * 32;
  int nBase = blockIdx.y * 128 + (wave >> 1) * 32;
  (void)M;

  f8 acc[2][2];
#pragma unroll
  for (int i = 0; i < 2; ++i)
#pragma unroll
    for (int j = 0; j < 2; ++j) acc[i][j] = zero8();

  for (int k0 = 0; k0 < K; k0 += 32) {
    h16 a0 = load_a_frag(A, K, mBase, k0, lane);
    h16 a1 = load_a_frag(A, K, mBase + 16, k0, lane);
    h16 b0 = load_b_frag(W, K, nBase, k0, lane);
    h16 b1 = load_b_frag(W, K, nBase + 16, k0, lane);
    acc[0][0] = wmma_f16(a0, b0, acc[0][0]);
    acc[0][1] = wmma_f16(a0, b1, acc[0][1]);
    acc[1][0] = wmma_f16(a1, b0, acc[1][0]);
    acc[1][1] = wmma_f16(a1, b1, acc[1][1]);
  }

#pragma unroll
  for (int i = 0; i < 2; ++i)
#pragma unroll
    for (int j = 0; j < 2; ++j) {
      int n = nBase + 16 * j + (lane & 15);
#pragma unroll
      for (int r = 0; r < 8; ++r) {
        int m   = mBase + 16 * i + r + ((lane >> 4) << 3);
        float v = acc[i][j][r] * scale;
        if (mode == 2) {
          out32[(size_t)m * N + n] = v;
        } else {
          int b = m / T_, t = m % T_;
          int h = n / D_, d = n % D_;
          if (mode == 0)
            out16[(((size_t)(b * H_ + h)) * T_ + t) * D_ + d] = (_Float16)v;
          else  // V^T
            out16[(((size_t)(b * H_ + h)) * D_ + d) * T_ + t] = (_Float16)v;
        }
      }
    }
}

// Online-softmax update for one 64-wide K chunk: s[4] are 16x16 score tiles in
// C-layout. Updates per-row (m,l) state, rescales o, converts P->f16 via LDS to
// A-fragments, accumulates o += P @ V (V^T layout, ld = T).
__device__ __forceinline__ void fa_update(f8 s[4], float* mst, float* lst, f8* o,
                                          _Float16* pb, const _Float16* VT,
                                          int kc, int lane) {
  float scl[8];
#pragma unroll
  for (int r = 0; r < 8; ++r) {
    float v = fmaxf(fmaxf(s[0][r], s[1][r]), fmaxf(s[2][r], s[3][r]));
#pragma unroll
    for (int off = 8; off >= 1; off >>= 1) v = fmaxf(v, __shfl_xor(v, off, 32));
    float nm = fmaxf(mst[r], v);
    scl[r]   = __expf(mst[r] - nm);
    mst[r]   = nm;
  }
#pragma unroll
  for (int r = 0; r < 8; ++r) {
    float rs = 0.f;
#pragma unroll
    for (int j = 0; j < 4; ++j) {
      float p = __expf(s[j][r] - mst[r]);
      s[j][r] = p;
      rs += p;
    }
#pragma unroll
    for (int off = 8; off >= 1; off >>= 1) rs += __shfl_xor(rs, off, 32);
    lst[r] = lst[r] * scl[r] + rs;
  }
#pragma unroll
  for (int dt = 0; dt < 4; ++dt)
#pragma unroll
    for (int r = 0; r < 8; ++r) o[dt][r] *= scl[r];

  // C-layout -> LDS (row-major 16x64, stride 72 halves)
#pragma unroll
  for (int j = 0; j < 4; ++j)
#pragma unroll
    for (int r = 0; r < 8; ++r) {
      int m = r + ((lane >> 4) << 3);
      pb[m * 72 + 16 * j + (lane & 15)] = (_Float16)s[j][r];
    }
  // LDS -> A fragments (same wave: DS ops are in-order)
  h16 pa0 = load_a_frag(pb, 72, 0, 0, lane);
  h16 pa1 = load_a_frag(pb, 72, 0, 32, lane);

#pragma unroll
  for (int dt = 0; dt < 4; ++dt) {
    h16 vb0 = load_b_frag(VT, T_, dt * 16, kc, lane);
    h16 vb1 = load_b_frag(VT, T_, dt * 16, kc + 32, lane);
    o[dt]   = wmma_f16(pa0, vb0, wmma_f16(pa1, vb1, o[dt]));
  }
}

__global__ __launch_bounds__(128) void diff_attn_wmma(
    const _Float16* __restrict__ q1, const _Float16* __restrict__ k1,
    const _Float16* __restrict__ q2, const _Float16* __restrict__ k2,
    const _Float16* __restrict__ vT, const float* __restrict__ subw,
    const float* __restrict__ lamp, _Float16* __restrict__ attnOut) {
  __shared__ __align__(16) _Float16 pbuf[4][16 * 72];
  int lane = threadIdx.x & 31;
  int wave = threadIdx.x >> 5;
  int h = blockIdx.y, b = blockIdx.z;
  int qRow = blockIdx.x * 64 + wave * 16;

  size_t headTD = (size_t)(b * H_ + h) * T_ * D_;
  const _Float16* Q1 = q1 + headTD;
  const _Float16* K1 = k1 + headTD;
  const _Float16* Q2 = q2 + headTD;
  const _Float16* K2 = k2 + headTD;
  const _Float16* VT = vT + headTD;  // [D,T] per head, same element count
  _Float16* pb = pbuf[wave];

  h16 qa1[2], qa2[2];
#pragma unroll
  for (int s = 0; s < 2; ++s) {
    qa1[s] = load_a_frag(Q1, D_, qRow, 32 * s, lane);
    qa2[s] = load_a_frag(Q2, D_, qRow, 32 * s, lane);
  }

  f8 o1[4], o2[4];
  float m1[8], l1[8], m2[8], l2[8];
#pragma unroll
  for (int dt = 0; dt < 4; ++dt) { o1[dt] = zero8(); o2[dt] = zero8(); }
#pragma unroll
  for (int r = 0; r < 8; ++r) {
    m1[r] = -1e30f; m2[r] = -1e30f; l1[r] = 0.f; l2[r] = 0.f;
  }

  for (int kc = 0; kc < T_; kc += 64) {
    f8 s[4];
#pragma unroll
    for (int j = 0; j < 4; ++j) {
      h16 kb0 = load_b_frag(K1, D_, kc + 16 * j, 0, lane);
      h16 kb1 = load_b_frag(K1, D_, kc + 16 * j, 32, lane);
      s[j] = wmma_f16(qa1[0], kb0, wmma_f16(qa1[1], kb1, zero8()));
    }
    fa_update(s, m1, l1, o1, pb, VT, kc, lane);
#pragma unroll
    for (int j = 0; j < 4; ++j) {
      h16 kb0 = load_b_frag(K2, D_, kc + 16 * j, 0, lane);
      h16 kb1 = load_b_frag(K2, D_, kc + 16 * j, 32, lane);
      s[j] = wmma_f16(qa2[0], kb0, wmma_f16(qa2[1], kb1, zero8()));
    }
    fa_update(s, m2, l2, o2, pb, VT, kc, lane);
  }

  float lam = *lamp;
#pragma unroll
  for (int r = 0; r < 8; ++r) {
    float inv1 = 1.f / l1[r];
    float inv2 = lam / l2[r];
    float ss = 0.f;
#pragma unroll
    for (int dt = 0; dt < 4; ++dt) {
      float a = o1[dt][r] * inv1 - o2[dt][r] * inv2;
      o1[dt][r] = a;
      ss += a * a;
    }
#pragma unroll
    for (int off = 8; off >= 1; off >>= 1) ss += __shfl_xor(ss, off, 32);
    float rms = rsqrtf(ss * (1.f / D_) + EPS_);
    int t = qRow + r + ((lane >> 4) << 3);
#pragma unroll
    for (int dt = 0; dt < 4; ++dt) {
      int d = dt * 16 + (lane & 15);
      float outv = o1[dt][r] * rms * subw[d] * ONE_MINUS_LI;
      attnOut[((size_t)b * T_ + t) * E_ + h * D_ + d] = (_Float16)outv;
    }
  }
}

extern "C" void kernel_launch(void* const* d_in, const int* in_sizes, int n_in,
                              void* d_out, int out_size, void* d_ws, size_t ws_size,
                              hipStream_t stream) {
  (void)in_sizes; (void)n_in; (void)out_size; (void)ws_size;
  const float* noisy_y = (const float*)d_in[0];
  const float* x       = (const float*)d_in[1];
  const float* Ws[6]   = {(const float*)d_in[2], (const float*)d_in[3],
                          (const float*)d_in[4], (const float*)d_in[5],
                          (const float*)d_in[6], (const float*)d_in[7]};
  const float* lq1  = (const float*)d_in[8];
  const float* lk1  = (const float*)d_in[9];
  const float* lq2  = (const float*)d_in[10];
  const float* lk2  = (const float*)d_in[11];
  const float* subw = (const float*)d_in[12];

  const size_t BTE = (size_t)B_ * T_ * E_;  // == B*H*T*D
  const size_t EE  = (size_t)E_ * E_;

  char* ws = (char*)d_ws;
  size_t off = 0;
  _Float16* yh = (_Float16*)(ws + off); off += BTE * 2;
  _Float16* xh = (_Float16*)(ws + off); off += BTE * 2;
  _Float16* wh[6];
  for (int i = 0; i < 6; ++i) { wh[i] = (_Float16*)(ws + off); off += EE * 2; }
  _Float16* q1h  = (_Float16*)(ws + off); off += BTE * 2;
  _Float16* k1h  = (_Float16*)(ws + off); off += BTE * 2;
  _Float16* q2h  = (_Float16*)(ws + off); off += BTE * 2;
  _Float16* k2h  = (_Float16*)(ws + off); off += BTE * 2;
  _Float16* vTh  = (_Float16*)(ws + off); off += BTE * 2;
  _Float16* atth = (_Float16*)(ws + off); off += BTE * 2;
  float* lamp    = (float*)(ws + off);

  cast_f32_f16<<<2048, 256, 0, stream>>>(noisy_y, yh, (long)BTE);
  cast_f32_f16<<<2048, 256, 0, stream>>>(x, xh, (long)BTE);
  for (int i = 0; i < 6; ++i)
    cast_f32_f16<<<1024, 256, 0, stream>>>(Ws[i], wh[i], (long)EE);
  lam_kernel<<<1, 32, 0, stream>>>(lq1, lk1, lq2, lk2, lamp);

  dim3 g(B_ * T_ / 64, E_ / 128), blk(256);
  const float qs = 0.125f;  // D^-0.5
  gemm_wmma<<<g, blk, 0, stream>>>(yh, wh[0], q1h, nullptr, B_ * T_, E_, E_, 0, qs);
  gemm_wmma<<<g, blk, 0, stream>>>(yh, wh[1], k1h, nullptr, B_ * T_, E_, E_, 0, 1.f);
  gemm_wmma<<<g, blk, 0, stream>>>(xh, wh[2], q2h, nullptr, B_ * T_, E_, E_, 0, qs);
  gemm_wmma<<<g, blk, 0, stream>>>(xh, wh[3], k2h, nullptr, B_ * T_, E_, E_, 0, 1.f);
  gemm_wmma<<<g, blk, 0, stream>>>(yh, wh[4], vTh, nullptr, B_ * T_, E_, E_, 1, 1.f);

  diff_attn_wmma<<<dim3(T_ / 64, H_, B_), 128, 0, stream>>>(
      q1h, k1h, q2h, k2h, vTh, subw, lamp, atth);

  gemm_wmma<<<g, blk, 0, stream>>>(atth, wh[5], nullptr, (float*)d_out,
                                   B_ * T_, E_, E_, 2, 1.f);
}
